// outer_product_42992622633050
// MI455X (gfx1250) — matrix operand order repaired and checked by reading, not compiled, not run on hardware
//
#include <hip/hip_runtime.h>
#include <hip/hip_bf16.h>

typedef __attribute__((ext_vector_type(16))) __bf16 v16bf;
typedef __attribute__((ext_vector_type(8)))  float  v8f;

#define DPAD     129
#define IJ       16641          // 129*129
#define K1TOT    2146689       // 129^3
#define KFULL    2146688       // K1TOT-1, multiple of 32
#define NSTEPS   67084         // KFULL/32
#define HID      128
#define BATCHN   128
#define KSUB     256           // k per LDS staging chunk
#define SUBSTEPS 8             // KSUB/32
#define LPITCH   264           // bf16 elems per LDS row (KSUB + 8 pad, 528B)
#define NWG      512

__global__ void zero_acc_kernel(float* __restrict__ y) {
    int i = blockIdx.x * blockDim.x + threadIdx.x;
    if (i < BATCHN * HID) y[i] = 0.0f;
}

__global__ void __launch_bounds__(256)
trilinear_gemm_kernel(const float* __restrict__ x,
                      const float* __restrict__ W1,
                      float* __restrict__ Yacc) {
    __shared__ __bf16 ldsU[BATCHN * LPITCH];   // 67584 bytes

    const int tid  = threadIdx.x;
    const int lane = tid & 31;
    const int wave = tid >> 5;

    // --- K-step range for this workgroup (uniform) ---
    const int spw  = (NSTEPS + NWG - 1) / NWG;
    const int sBeg = blockIdx.x * spw;
    const int sEnd = min(NSTEPS, sBeg + spw);
    if (sBeg >= sEnd) return;

    // --- fill-role constants: thread fills row fb, cols fc0, fc0+2, ... ---
    const int fb  = tid & 127;
    const int fc0 = tid >> 7;
    const float* xb = x + fb * 384;     // x[fb, 0:3, 0:128]

    // --- compute-role constants (wave w handles output cols h = 16w..16w+15) ---
    const int nIdx = lane & 15;
    const int half = lane >> 4;                     // 0: K 0-15, 1: K 16-31 of B tile
    const int hcol = wave * 16 + nIdx;
    const size_t wRowBase = (size_t)hcol * K1TOT + (size_t)half * 16;

    v8f acc[8];
    #pragma unroll
    for (int mt = 0; mt < 8; ++mt)
        acc[mt] = (v8f){0.f,0.f,0.f,0.f,0.f,0.f,0.f,0.f};

    for (int s0 = sBeg; s0 < sEnd; s0 += SUBSTEPS) {
        const int nst   = min(SUBSTEPS, sEnd - s0);
        const int kbase = s0 * 32;
        const int ncols = nst * 32;

        // ---- generate U tile (bf16) into LDS: U[b,c] = a_i * v_j * t_kk ----
        for (int c = fc0; c < ncols; c += 2) {
            unsigned k  = (unsigned)(kbase + c);
            unsigned i  = k / IJ;
            unsigned r  = k - i * IJ;
            unsigned j  = r / DPAD;
            unsigned kk = r - j * DPAD;
            float av = (i  == 0) ? 1.0f : xb[i - 1];
            float vv = (j  == 0) ? 1.0f : xb[128 + j - 1];
            float tv = (kk == 0) ? 1.0f : xb[256 + kk - 1];
            ldsU[fb * LPITCH + c] = (__bf16)(av * vv * tv);
        }
        __syncthreads();

        // ---- WMMA over this chunk ----
        for (int ks = 0; ks < nst; ++ks) {
            const float* wp = W1 + wRowBase + (size_t)(kbase + ks * 32);
            if (kbase + ks * 32 + KSUB < KFULL)
                __builtin_prefetch(wp + KSUB, 0, 1);

            // B operand: 32x16 bf16, lane holds 16 consecutive k of row h
            float bf[16];
            #pragma unroll
            for (int e = 0; e < 16; ++e) bf[e] = __builtin_nontemporal_load(wp + e);
            v16bf bB;
            #pragma unroll
            for (int e = 0; e < 16; ++e) bB[e] = (__bf16)bf[e];

            #pragma unroll
            for (int mt = 0; mt < 8; ++mt) {
                // A operand: 16x32 bf16 from LDS
                // lane<16: bytes +0/+16 ; lane>=16: bytes +32/+48
                const __bf16* ap = ldsU + (mt * 16 + nIdx) * LPITCH
                                        + ks * 32 + half * 16;
                union { uint4 u4[2]; v16bf v; } A;
                A.u4[0] = *(const uint4*)(ap);
                A.u4[1] = *(const uint4*)(ap + 8);
                acc[mt] = __builtin_amdgcn_wmma_f32_16x16x32_bf16(
                    false, A.v, false, bB, (short)0, acc[mt], false, false);
            }
        }
        __syncthreads();
    }

    // ---- accumulate partial 128x128 tile into global Yacc ----
    #pragma unroll
    for (int mt = 0; mt < 8; ++mt) {
        #pragma unroll
        for (int r = 0; r < 8; ++r) {
            int row = mt * 16 + r + half * 8;   // batch index
            unsafeAtomicAdd(&Yacc[row * HID + hcol], acc[mt][r]);
        }
    }
}

__global__ void __launch_bounds__(128)
mlp_epilogue_kernel(const float* __restrict__ x,  const float* __restrict__ W1,
                    const float* __restrict__ b1, const float* __restrict__ W2,
                    const float* __restrict__ b2, const float* __restrict__ W3,
                    const float* __restrict__ b3, const float* __restrict__ Yacc,
                    float* __restrict__ out) {
    if (blockIdx.x > 0) {
        // copy x (second tuple element) into out[1024 .. 1024+49151]
        int idx = (blockIdx.x - 1) * 128 + threadIdx.x;   // 0..49151
        out[1024 + idx] = x[idx];
        return;
    }
    const int b = threadIdx.x;          // 0..127
    const float* xb = x + b * 384;

    // leftover einsum term: k = K1-1  (i=j=kk=128)
    float ulast = xb[127] * xb[255] * xb[383];

    float l1[HID];
    for (int h = 0; h < HID; ++h) {
        float s = Yacc[b * HID + h]
                + ulast * W1[(size_t)h * K1TOT + (K1TOT - 1)]
                + b1[h];
        l1[h] = fmaxf(s, 0.0f);
    }
    float l2[HID];
    for (int h2 = 0; h2 < HID; ++h2) {
        float s = b2[h2];
        for (int h = 0; h < HID; ++h) s += W2[h2 * HID + h] * l1[h];
        l2[h2] = fmaxf(s, 0.0f);
    }
    float y3[8];
    float m = -1e30f;
    for (int o = 0; o < 8; ++o) {
        float s = b3[o];
        for (int h = 0; h < HID; ++h) s += W3[o * HID + h] * l2[h];
        y3[o] = 1.0f / (1.0f + __expf(-s));
        m = fmaxf(m, y3[o]);
    }
    float denom = 0.0f;
    for (int o = 0; o < 8; ++o) { y3[o] = __expf(y3[o] - m); denom += y3[o]; }
    for (int o = 0; o < 8; ++o) out[b * 8 + o] = y3[o] / denom;
}

extern "C" void kernel_launch(void* const* d_in, const int* in_sizes, int n_in,
                              void* d_out, int out_size, void* d_ws, size_t ws_size,
                              hipStream_t stream) {
    const float* x  = (const float*)d_in[0];
    const float* W1 = (const float*)d_in[1];
    const float* b1 = (const float*)d_in[2];
    const float* W2 = (const float*)d_in[3];
    const float* b2 = (const float*)d_in[4];
    const float* W3 = (const float*)d_in[5];
    const float* b3 = (const float*)d_in[6];
    float* out  = (float*)d_out;
    float* Yacc = (float*)d_ws;          // 128*128 f32 = 64 KB

    zero_acc_kernel<<<(BATCHN * HID + 255) / 256, 256, 0, stream>>>(Yacc);
    trilinear_gemm_kernel<<<NWG, 256, 0, stream>>>(x, W1, Yacc);
    // 1 MLP block + 384 copy blocks (384*128 = 49152 = |x|)
    mlp_epilogue_kernel<<<1 + 384, 128, 0, stream>>>(x, W1, b1, W2, b2, W3, b3,
                                                     Yacc, out);
}
